// MultiHorizonXSLTM_35931696398361
// MI455X (gfx1250) — compile-verified, matrix operand order checked by reading, not run on hardware
//
#include <hip/hip_runtime.h>
#include <hip/hip_bf16.h>

// ---------------- model constants ----------------
constexpr int CB = 32, CS = 512, CDIN = 128, CD = 512, CINNER = 1024;
constexpr int CNH_M = 4, CDH_M = 256, CNH_S = 4, CDH_S = 128, CKCONV = 4, CFF = 704;
constexpr size_t CBS = (size_t)CB * CS; // 16384 rows

typedef __attribute__((ext_vector_type(16))) __bf16 v16bf;
typedef __attribute__((ext_vector_type(8)))  float  v8f;

// ---------------- device helpers ----------------
__device__ __forceinline__ float sigm_(float x) { return 1.f / (1.f + __expf(-x)); }
__device__ __forceinline__ float silu_(float x) { return x * sigm_(x); }
__device__ __forceinline__ float logsig_(float x) {
  float m = fminf(x, 0.f);
  return m - log1pf(__expf(-fabsf(x)));
}

// bf16 WMMA fragment K-pair base for VGPR v, half = lane>>4
__device__ __forceinline__ int frag_k(int v, int half) {
  return (v < 4 ? 2 * v : 16 + 2 * (v - 4)) + half * 8;
}

// ---------------- elementwise convert ----------------
__global__ void k_f32_to_bf16(const float* __restrict__ src, __bf16* __restrict__ dst, int n) {
  int i = blockIdx.x * blockDim.x + threadIdx.x;
  if (i < n) dst[i] = (__bf16)src[i];
}

// ---------------- generic bf16 WMMA GEMM ----------------
// C[M,N](f32) = A[M,K](bf16,lda) * W[N,K](bf16,ldw)^T + bias?[N] + residual?[M,N]
#define GT_BM 64
#define GT_BN 64
#define GT_BK 32
__global__ __launch_bounds__(256)
void k_gemm_bf16(const __bf16* __restrict__ A, int lda,
                 const __bf16* __restrict__ W, int ldw,
                 const float* __restrict__ bias,
                 const float* __restrict__ residual, int ldr,
                 float* __restrict__ C, int ldc,
                 int M, int N, int K) {
  __shared__ alignas(16) __bf16 sA[GT_BM * GT_BK];
  __shared__ alignas(16) __bf16 sW[GT_BN * GT_BK];
  const int tid = threadIdx.x;
  const int wave = tid >> 5, lane = tid & 31;
  const int wm = wave >> 1, wn = wave & 1; // 4(M) x 2(N) waves
  const int m0 = blockIdx.y * GT_BM;
  const int n0 = blockIdx.x * GT_BN;
  const int half = lane >> 4, l16 = lane & 15;

  v8f acc0 = {0.f, 0.f, 0.f, 0.f, 0.f, 0.f, 0.f, 0.f};
  v8f acc1 = acc0;

  const int le = tid * 8;
  const int lr = le >> 5, lc = le & 31;

  for (int k0 = 0; k0 < K; k0 += GT_BK) {
    // stage 64x32 bf16 tiles of A and W (b128 per thread)
    *(uint4*)&sA[le] = *(const uint4*)&A[(size_t)(m0 + lr) * lda + k0 + lc];
    *(uint4*)&sW[le] = *(const uint4*)&W[(size_t)(n0 + lr) * ldw + k0 + lc];
    if (k0 + GT_BK < K) { // hint next tiles into cache (global_prefetch_b8)
      __builtin_prefetch(&A[(size_t)(m0 + lr) * lda + k0 + GT_BK + lc], 0, 1);
      __builtin_prefetch(&W[(size_t)(n0 + lr) * ldw + k0 + GT_BK + lc], 0, 1);
    }
    __syncthreads();

    v16bf af;
    {
      const __bf16* base = &sA[(wm * 16 + l16) * GT_BK];
      #pragma unroll
      for (int v = 0; v < 8; ++v)
        ((unsigned*)&af)[v] = *(const unsigned*)&base[frag_k(v, half)];
    }
    #pragma unroll
    for (int sub = 0; sub < 2; ++sub) {
      v16bf bfr;
      const __bf16* base = &sW[(wn * 32 + sub * 16 + l16) * GT_BK];
      #pragma unroll
      for (int v = 0; v < 8; ++v)
        ((unsigned*)&bfr)[v] = *(const unsigned*)&base[frag_k(v, half)];
      if (sub == 0)
        acc0 = __builtin_amdgcn_wmma_f32_16x16x32_bf16(false, af, false, bfr, (short)0, acc0, false, false);
      else
        acc1 = __builtin_amdgcn_wmma_f32_16x16x32_bf16(false, af, false, bfr, (short)0, acc1, false, false);
    }
    __syncthreads();
  }

  #pragma unroll
  for (int r = 0; r < 8; ++r) {
    int mi = m0 + wm * 16 + half * 8 + r;
    {
      int ni = n0 + wn * 32 + l16;
      float v = acc0[r];
      if (bias) v += bias[ni];
      if (residual) v += residual[(size_t)mi * ldr + ni];
      C[(size_t)mi * ldc + ni] = v;
    }
    {
      int ni = n0 + wn * 32 + 16 + l16;
      float v = acc1[r];
      if (bias) v += bias[ni];
      if (residual) v += residual[(size_t)mi * ldr + ni];
      C[(size_t)mi * ldc + ni] = v;
    }
  }
}

// ---------------- layernorm (D=512) ----------------
__global__ __launch_bounds__(256)
void k_layernorm(const float* __restrict__ x, const float* __restrict__ w, float* __restrict__ out) {
  size_t row = blockIdx.x;
  int t = threadIdx.x;
  float v0 = x[row * 512 + t], v1 = x[row * 512 + 256 + t];
  __shared__ float red[256];
  red[t] = v0 + v1;
  __syncthreads();
  for (int off = 128; off > 0; off >>= 1) { if (t < off) red[t] += red[t + off]; __syncthreads(); }
  float mu = red[0] * (1.f / 512.f);
  __syncthreads();
  float d0 = v0 - mu, d1 = v1 - mu;
  red[t] = d0 * d0 + d1 * d1;
  __syncthreads();
  for (int off = 128; off > 0; off >>= 1) { if (t < off) red[t] += red[t + off]; __syncthreads(); }
  float inv = rsqrtf(red[0] * (1.f / 512.f) + 1e-5f);
  out[row * 512 + t] = d0 * inv * w[t];
  out[row * 512 + 256 + t] = d1 * inv * w[256 + t];
}

// ---------------- depthwise causal conv (K=4) + SiLU ----------------
__global__ void k_conv_silu(const float* __restrict__ x, const float* __restrict__ w,
                            const float* __restrict__ bias, float* __restrict__ out,
                            int F, int total) {
  int idx = blockIdx.x * blockDim.x + threadIdx.x;
  if (idx >= total) return;
  int f = idx % F;
  int s = (idx / F) % CS;
  int b = idx / (F * CS);
  float acc = bias[f];
  #pragma unroll
  for (int t = 0; t < CKCONV; ++t) {
    int sp = s - (CKCONV - 1) + t;
    if (sp >= 0) acc += x[((size_t)b * CS + sp) * F + f] * w[f * CKCONV + t];
  }
  out[idx] = silu_(acc);
}

// ---------------- block-diagonal 4x4 headwise -> bf16 ----------------
__global__ void k_headwise4_bf(const float* __restrict__ x, const float* __restrict__ w,
                               __bf16* __restrict__ out, int total) {
  int idx = blockIdx.x * blockDim.x + threadIdx.x;
  if (idx >= total) return;
  int c = idx % CINNER;
  int hs = c >> 2, o = c & 3;
  size_t row = idx / CINNER;
  const float* xr = x + row * CINNER + (hs << 2);
  const float* wr = w + ((size_t)hs * 4 + o) * 4; // (256,4,4)[h][o][i]
  float acc = xr[0] * wr[0] + xr[1] * wr[1] + xr[2] * wr[2] + xr[3] * wr[3];
  out[idx] = (__bf16)acc;
}

// ---------------- mLSTM input/forget gate preactivations ----------------
__global__ __launch_bounds__(128)
void k_gates(const __bf16* __restrict__ qb, const __bf16* __restrict__ kb,
             const __bf16* __restrict__ vb,
             const float* __restrict__ wig, const float* __restrict__ big,
             const float* __restrict__ wfg, const float* __restrict__ bfg,
             float* __restrict__ ig, float* __restrict__ fg) {
  size_t row = blockIdx.x; // b*S + s
  int b = (int)(row / CS), s = (int)(row % CS);
  int tid = threadIdx.x;
  __shared__ float red[8][128];
  float acc[8] = {0, 0, 0, 0, 0, 0, 0, 0};
  for (int c = tid; c < CINNER; c += 128) {
    float qv = (float)qb[row * CINNER + c];
    float kv = (float)kb[row * CINNER + c];
    float vv = (float)vb[row * CINNER + c];
    #pragma unroll
    for (int h = 0; h < 4; ++h) {
      const float* wi = wig + (size_t)h * 3 * CINNER;
      const float* wf = wfg + (size_t)h * 3 * CINNER;
      acc[h]     += qv * wi[c] + kv * wi[CINNER + c] + vv * wi[2 * CINNER + c];
      acc[4 + h] += qv * wf[c] + kv * wf[CINNER + c] + vv * wf[2 * CINNER + c];
    }
  }
  #pragma unroll
  for (int g = 0; g < 8; ++g) red[g][tid] = acc[g];
  __syncthreads();
  for (int off = 64; off > 0; off >>= 1) {
    if (tid < off) {
      #pragma unroll
      for (int g = 0; g < 8; ++g) red[g][tid] += red[g][tid + off];
    }
    __syncthreads();
  }
  if (tid < 4) {
    ig[((size_t)b * 4 + tid) * CS + s] = red[tid][0] + big[tid];
    fg[((size_t)b * 4 + tid) * CS + s] = red[4 + tid][0] + bfg[tid];
  }
}

// ---------------- mLSTM stabilizer scan (cumsum logsig + prefix max) ----------------
__global__ void k_mlstm_scan(const float* __restrict__ ig, const float* __restrict__ fg,
                             float* __restrict__ logfc, float* __restrict__ mrow) {
  int bh = blockIdx.x * blockDim.x + threadIdx.x;
  if (bh >= CB * CNH_M) return;
  const float* f = fg + (size_t)bh * CS;
  const float* i_ = ig + (size_t)bh * CS;
  float lc = 0.f, pm = -INFINITY;
  for (int s = 0; s < CS; ++s) {
    lc += logsig_(f[s]);
    logfc[(size_t)bh * CS + s] = lc;
    pm = fmaxf(pm, i_[s] - lc);
    mrow[(size_t)bh * CS + s] = lc + pm; // m_i
  }
}

// ---------------- mLSTM parallel attention (causal, stabilized) ----------------
// grid: (S/16 query tiles, B*NH); one wave per block; out (B,NH,S,DH=256) f32
__global__ __launch_bounds__(32)
void k_mlstm_attn(const __bf16* __restrict__ qb, const __bf16* __restrict__ kb,
                  const __bf16* __restrict__ vb,
                  const float* __restrict__ igv, const float* __restrict__ logfc,
                  const float* __restrict__ mrow, float* __restrict__ outp) {
  const int bh = blockIdx.y;
  const int qt = blockIdx.x;
  const int lane = threadIdx.x;
  const int half = lane >> 4, l16 = lane & 15;
  const __bf16* q = qb + (size_t)bh * CS * CDH_M;
  const __bf16* k = kb + (size_t)bh * CS * CDH_M;
  const __bf16* v = vb + (size_t)bh * CS * CDH_M;
  const float* ig_ = igv + (size_t)bh * CS;
  const float* lf_ = logfc + (size_t)bh * CS;
  const float* mr_ = mrow + (size_t)bh * CS;
  const int qrow0 = qt * 16;

  __shared__ alignas(16) __bf16 sP[16 * 32];

  // preload all 8 A-fragments of the 16x256 query tile
  v16bf qf[8];
  #pragma unroll
  for (int kk = 0; kk < 8; ++kk) {
    const __bf16* base = q + (size_t)(qrow0 + l16) * CDH_M + kk * 32;
    #pragma unroll
    for (int vv = 0; vv < 8; ++vv)
      ((unsigned*)&qf[kk])[vv] = *(const unsigned*)&base[frag_k(vv, half)];
  }
  float mi[8], lfi[8];
  #pragma unroll
  for (int rr = 0; rr < 8; ++rr) {
    int i = qrow0 + half * 8 + rr;
    mi[rr] = mr_[i];
    lfi[rr] = lf_[i];
  }
  const v8f vzero = {0.f, 0.f, 0.f, 0.f, 0.f, 0.f, 0.f, 0.f};
  v8f oacc[16];
  #pragma unroll
  for (int t = 0; t < 16; ++t) oacc[t] = vzero;
  float rsum[8] = {0, 0, 0, 0, 0, 0, 0, 0};
  const float scale = 0.0625f; // rsqrt(256)

  const int npair = (qt + 2) >> 1; // pairs of 16-key tiles (K=32 per P@V WMMA)
  for (int jp = 0; jp < npair; ++jp) {
    const int jbase = jp * 32;
    #pragma unroll
    for (int sub = 0; sub < 2; ++sub) {
      const int jb = jbase + sub * 16;
      v8f s = vzero;
      #pragma unroll
      for (int kk = 0; kk < 8; ++kk) {
        v16bf kf;
        const __bf16* base = k + (size_t)(jb + l16) * CDH_M + kk * 32;
        #pragma unroll
        for (int vv = 0; vv < 8; ++vv)
          ((unsigned*)&kf)[vv] = *(const unsigned*)&base[frag_k(vv, half)];
        s = __builtin_amdgcn_wmma_f32_16x16x32_bf16(false, qf[kk], false, kf, (short)0, s, false, false);
      }
      const int j = jb + l16;
      const float lgj = lf_[j], igj = ig_[j];
      #pragma unroll
      for (int rr = 0; rr < 8; ++rr) {
        int i = qrow0 + half * 8 + rr;
        float p = 0.f;
        if (j <= i) p = s[rr] * scale * __expf(lfi[rr] - lgj + igj - mi[rr]);
        rsum[rr] += p;
        sP[(half * 8 + rr) * 32 + sub * 16 + l16] = (__bf16)p;
      }
    }
    __syncthreads();
    v16bf pf;
    {
      const __bf16* base = &sP[l16 * 32];
      #pragma unroll
      for (int vv = 0; vv < 8; ++vv)
        ((unsigned*)&pf)[vv] = *(const unsigned*)&base[frag_k(vv, half)];
    }
    #pragma unroll
    for (int nb = 0; nb < 16; ++nb) {
      v16bf vf;
      #pragma unroll
      for (int vv = 0; vv < 8; ++vv) {
        int kki = frag_k(vv, half);
        const __bf16* e0 = v + (size_t)(jbase + kki) * CDH_M + nb * 16 + l16;
        unsigned lo = *(const unsigned short*)e0;
        unsigned hi = *(const unsigned short*)(e0 + CDH_M);
        ((unsigned*)&vf)[vv] = lo | (hi << 16);
      }
      oacc[nb] = __builtin_amdgcn_wmma_f32_16x16x32_bf16(false, pf, false, vf, (short)0, oacc[nb], false, false);
    }
    __syncthreads();
  }
  // row-sum butterfly within each 16-lane half
  #pragma unroll
  for (int rr = 0; rr < 8; ++rr) {
    float r = rsum[rr];
    r += __shfl_xor(r, 1, 32);
    r += __shfl_xor(r, 2, 32);
    r += __shfl_xor(r, 4, 32);
    r += __shfl_xor(r, 8, 32);
    rsum[rr] = r;
  }
  #pragma unroll
  for (int rr = 0; rr < 8; ++rr) {
    int i = qrow0 + half * 8 + rr;
    float inv = 1.f / (fmaxf(fabsf(rsum[rr]), __expf(-mi[rr])) + 1e-6f);
    #pragma unroll
    for (int nb = 0; nb < 16; ++nb)
      outp[((size_t)bh * CS + i) * CDH_M + nb * 16 + l16] = oacc[nb][rr] * inv;
  }
}

// ---------------- mLSTM out-head LN + skip + z-gating -> bf16 ----------------
__global__ __launch_bounds__(256)
void k_mh_gate(const float* __restrict__ att, const float* __restrict__ xc,
               const float* __restrict__ z, const float* __restrict__ onorm,
               const float* __restrict__ skip, __bf16* __restrict__ gbf) {
  int blk = blockIdx.x;
  int h = blk & 3;
  size_t row = (size_t)(blk >> 2); // b*S+s
  int b = (int)(row / CS), s = (int)(row % CS);
  int d = threadIdx.x;
  float val = att[(((size_t)b * 4 + h) * CS + s) * CDH_M + d];
  __shared__ float red[256];
  red[d] = val;
  __syncthreads();
  for (int off = 128; off > 0; off >>= 1) { if (d < off) red[d] += red[d + off]; __syncthreads(); }
  float mu = red[0] * (1.f / 256.f);
  __syncthreads();
  float dv = val - mu;
  red[d] = dv * dv;
  __syncthreads();
  for (int off = 128; off > 0; off >>= 1) { if (d < off) red[d] += red[d + off]; __syncthreads(); }
  float hn = dv * rsqrtf(red[0] * (1.f / 256.f) + 1e-5f);
  int col = h * CDH_M + d;
  float hs = hn * onorm[col] + skip[col] * xc[row * CINNER + col];
  float zz = z[row * CINNER + col];
  gbf[row * CINNER + col] = (__bf16)(hs * silu_(zz));
}

// ---------------- sLSTM recurrent scan ----------------
// grid: B*NH_S blocks, 128 threads; wx (B*S, NH, 512); R (NH,128,512); y (B,S,512)
__global__ __launch_bounds__(128)
void k_slstm_scan(const float* __restrict__ wx, const float* __restrict__ R,
                  const float* __restrict__ bias, float* __restrict__ y) {
  int blk = blockIdx.x;
  int b = blk >> 2, h = blk & 3;
  int e = threadIdx.x;
  __shared__ float hsh[128];
  hsh[e] = 0.f;
  float c = 0.f, n = 0.f, m = 0.f;
  const float* Rh = R + (size_t)h * 128 * 512;
  float bi = bias[h * 512 + e], bff = bias[h * 512 + 128 + e];
  float bz = bias[h * 512 + 256 + e], bo = bias[h * 512 + 384 + e];
  __syncthreads();
  for (int s = 0; s < CS; ++s) {
    const float* wxr = wx + ((size_t)b * CS + s) * 2048 + (size_t)h * 512;
    float ri = wxr[e] + bi, rf = wxr[128 + e] + bff;
    float rz = wxr[256 + e] + bz, ro = wxr[384 + e] + bo;
    for (int d = 0; d < 128; ++d) {
      float hv = hsh[d];
      const float* rr = Rh + (size_t)d * 512;
      ri += hv * rr[e];
      rf += hv * rr[128 + e];
      rz += hv * rr[256 + e];
      ro += hv * rr[384 + e];
    }
    float lfpm = m + logsig_(rf);
    float mnew = (n == 0.f) ? ri : fmaxf(ri, lfpm);
    float og = sigm_(ro);
    float igt = __expf(ri - mnew);
    float fgt = __expf(lfpm - mnew);
    c = fgt * c + igt * tanhf(rz);
    n = fgt * n + igt;
    m = mnew;
    float yv = og * c / n;
    y[((size_t)b * CS + s) * 512 + h * 128 + e] = yv;
    __syncthreads();
    hsh[e] = yv;
    __syncthreads();
  }
}

// ---------------- sLSTM per-head groupnorm + residual into h ----------------
__global__ __launch_bounds__(128)
void k_slstm_norm_res(const float* __restrict__ y, const float* __restrict__ gn,
                      float* __restrict__ h_io) {
  int blk = blockIdx.x;
  int hh = blk & 3;
  size_t row = (size_t)(blk >> 2);
  int e = threadIdx.x;
  float val = y[row * 512 + hh * 128 + e];
  __shared__ float red[128];
  red[e] = val;
  __syncthreads();
  for (int off = 64; off > 0; off >>= 1) { if (e < off) red[e] += red[e + off]; __syncthreads(); }
  float mu = red[0] * (1.f / 128.f);
  __syncthreads();
  float dv = val - mu;
  red[e] = dv * dv;
  __syncthreads();
  for (int off = 64; off > 0; off >>= 1) { if (e < off) red[e] += red[e + off]; __syncthreads(); }
  int col = hh * 128 + e;
  h_io[row * 512 + col] += dv * rsqrtf(red[0] * (1.f / 128.f) + 1e-5f) * gn[col];
}

// ---------------- FFN activation: gelu(gate)*u -> bf16 ----------------
__global__ void k_ffn_act(const float* __restrict__ up, __bf16* __restrict__ actbf, int total) {
  int i = blockIdx.x * blockDim.x + threadIdx.x;
  if (i >= total) return;
  int cc = i % CFF;
  size_t row = (size_t)(i / CFF);
  float g = up[row * (2 * CFF) + cc], u = up[row * (2 * CFF) + CFF + cc];
  float ge = 0.5f * g * (1.f + erff(g * 0.70710678f));
  actbf[i] = (__bf16)(ge * u);
}

// ---------------- final head: sigmoid(h_last @ w_fc^T + b_fc) ----------------
__global__ __launch_bounds__(256)
void k_final(const float* __restrict__ hn, const float* __restrict__ wfc,
             const float* __restrict__ bfc, float* __restrict__ out) {
  int b = blockIdx.x;
  int t = threadIdx.x;
  const float* xr = hn + ((size_t)b * CS + (CS - 1)) * 512;
  __shared__ float red[3][256];
  float a0 = 0, a1 = 0, a2 = 0;
  for (int d = t; d < 512; d += 256) {
    float xv = xr[d];
    a0 += xv * wfc[d];
    a1 += xv * wfc[512 + d];
    a2 += xv * wfc[1024 + d];
  }
  red[0][t] = a0; red[1][t] = a1; red[2][t] = a2;
  __syncthreads();
  for (int off = 128; off > 0; off >>= 1) {
    if (t < off) {
      red[0][t] += red[0][t + off];
      red[1][t] += red[1][t + off];
      red[2][t] += red[2][t + off];
    }
    __syncthreads();
  }
  if (t < 3) out[b * 3 + t] = sigm_(red[t][0] + bfc[t]);
}

// =====================================================================
extern "C" void kernel_launch(void* const* d_in, const int* in_sizes, int n_in,
                              void* d_out, int out_size, void* d_ws, size_t ws_size,
                              hipStream_t stream) {
  (void)in_sizes; (void)n_in; (void)out_size; (void)ws_size;
  const float* x        = (const float*)d_in[0];
  const float* w_in     = (const float*)d_in[1];
  const float* b_in     = (const float*)d_in[2];
  const float* m_ln     = (const float*)d_in[3];
  const float* m_wup    = (const float*)d_in[4];
  const float* m_conv_w = (const float*)d_in[5];
  const float* m_conv_b = (const float*)d_in[6];
  const float* m_wq     = (const float*)d_in[7];
  const float* m_wk     = (const float*)d_in[8];
  const float* m_wv     = (const float*)d_in[9];
  const float* m_wig    = (const float*)d_in[10];
  const float* m_big    = (const float*)d_in[11];
  const float* m_wfg    = (const float*)d_in[12];
  const float* m_bfg    = (const float*)d_in[13];
  const float* m_outnorm= (const float*)d_in[14];
  const float* m_skip   = (const float*)d_in[15];
  const float* m_wdown  = (const float*)d_in[16];
  const float* s_ln     = (const float*)d_in[17];
  const float* s_conv_w = (const float*)d_in[18];
  const float* s_conv_b = (const float*)d_in[19];
  const float* s_w[4]   = {(const float*)d_in[20], (const float*)d_in[21],
                           (const float*)d_in[22], (const float*)d_in[23]};
  const float* s_R      = (const float*)d_in[24];
  const float* s_bv     = (const float*)d_in[25];
  const float* s_gn     = (const float*)d_in[26];
  const float* s_ln2    = (const float*)d_in[27];
  const float* f_wup    = (const float*)d_in[28];
  const float* f_wdown  = (const float*)d_in[29];
  const float* post_ln  = (const float*)d_in[30];
  const float* w_fc     = (const float*)d_in[31];
  const float* b_fc     = (const float*)d_in[32];
  float* out = (float*)d_out;

  // ---------------- workspace layout ----------------
  char* ws = (char*)d_ws;
  size_t off = 0;
  auto take = [&](size_t bytes) {
    size_t o = off;
    off += (bytes + 255) & ~(size_t)255;
    return o;
  };
  // persistent
  float*  h_buf   = (float*)(ws + take(CBS * 512 * 4));
  float*  ln_buf  = (float*)(ws + take(CBS * 512 * 4));
  __bf16* lnbf    = (__bf16*)(ws + take(CBS * 512 * 2));
  __bf16* xbf     = (__bf16*)(ws + take(CBS * CDIN * 2));
  __bf16* w_in_bf = (__bf16*)(ws + take((size_t)512 * 128 * 2));
  __bf16* wup_bf  = (__bf16*)(ws + take((size_t)2048 * 512 * 2));
  __bf16* wdown_bf= (__bf16*)(ws + take((size_t)512 * 1024 * 2));
  __bf16* sw_bf[4];
  for (int i = 0; i < 4; ++i) sw_bf[i] = (__bf16*)(ws + take((size_t)4 * 128 * 128 * 2));
  __bf16* fup_bf  = (__bf16*)(ws + take((size_t)1408 * 512 * 2));
  __bf16* fdn_bf  = (__bf16*)(ws + take((size_t)512 * 704 * 2));
  float*  ig_buf  = (float*)(ws + take((size_t)CB * 4 * CS * 4));
  float*  fg_buf  = (float*)(ws + take((size_t)CB * 4 * CS * 4));
  float*  lfc_buf = (float*)(ws + take((size_t)CB * 4 * CS * 4));
  float*  mrw_buf = (float*)(ws + take((size_t)CB * 4 * CS * 4));
  // stage arena (regions reused across the three sequential stages)
  size_t A0 = take((size_t)336 * 1024 * 1024);
  const size_t F1 = CBS * 1024 * 4; // 67,108,864
  // mLSTM stage views
  float*  z_buf  = (float*)(ws + A0);
  float*  xc_buf = (float*)(ws + A0 + F1);
  float*  xm_buf = (float*)(ws + A0 + 2 * F1);        // reused later as att
  float*  att    = xm_buf;
  __bf16* qb     = (__bf16*)(ws + A0 + 3 * F1);
  __bf16* kb     = (__bf16*)(ws + A0 + 3 * F1 + F1 / 2);
  __bf16* vb     = (__bf16*)(ws + A0 + 3 * F1 + 2 * (F1 / 2));
  __bf16* gbf    = (__bf16*)(ws + A0 + 3 * F1 + 3 * (F1 / 2));
  // sLSTM stage views (overlap mLSTM regions; safe: stages are sequential)
  float*  xc2    = (float*)(ws + A0);
  __bf16* xc2bf  = (__bf16*)(ws + A0 + CBS * 512 * 4);
  float*  wx_buf = (float*)(ws + A0 + CBS * 512 * 4 + CBS * 512 * 2 + 256);
  float*  y_buf  = (float*)(ws + A0 + CBS * 512 * 4 + CBS * 512 * 2 + 256 + CBS * 2048 * 4);
  // FFN stage views
  float*  ffup   = (float*)(ws + A0);
  __bf16* ffbf   = (__bf16*)(ws + A0 + CBS * 1408 * 4);

  const int TB = 256;
  auto cgrid = [&](size_t n) { return dim3((unsigned)((n + TB - 1) / TB)); };
  auto conv_bf = [&](const float* src, __bf16* dst, size_t n) {
    k_f32_to_bf16<<<cgrid(n), TB, 0, stream>>>(src, dst, (int)n);
  };
  auto gemm = [&](const __bf16* A, int lda, const __bf16* W, int ldw,
                  const float* bias, const float* res, int ldr,
                  float* C, int ldc, int M, int N, int K) {
    k_gemm_bf16<<<dim3(N / 64, M / 64), 256, 0, stream>>>(A, lda, W, ldw, bias, res, ldr, C, ldc, M, N, K);
  };

  // ---------------- weight + input conversion ----------------
  conv_bf(x, xbf, CBS * CDIN);
  conv_bf(w_in, w_in_bf, (size_t)512 * 128);
  conv_bf(m_wup, wup_bf, (size_t)2048 * 512);
  conv_bf(m_wdown, wdown_bf, (size_t)512 * 1024);
  for (int i = 0; i < 4; ++i) conv_bf(s_w[i], sw_bf[i], (size_t)4 * 128 * 128);
  conv_bf(f_wup, fup_bf, (size_t)1408 * 512);
  conv_bf(f_wdown, fdn_bf, (size_t)512 * 704);

  const int M = (int)CBS;

  // ---------------- input projection ----------------
  gemm(xbf, CDIN, w_in_bf, CDIN, b_in, nullptr, 0, h_buf, 512, M, 512, CDIN);

  // ---------------- mLSTM block ----------------
  k_layernorm<<<dim3(M), 256, 0, stream>>>(h_buf, m_ln, ln_buf);
  conv_bf(ln_buf, lnbf, CBS * 512);
  gemm(lnbf, 512, wup_bf, 512, nullptr, nullptr, 0, xm_buf, 1024, M, 1024, 512);
  gemm(lnbf, 512, wup_bf + (size_t)1024 * 512, 512, nullptr, nullptr, 0, z_buf, 1024, M, 1024, 512);
  k_conv_silu<<<cgrid(CBS * 1024), TB, 0, stream>>>(xm_buf, m_conv_w, m_conv_b, xc_buf, 1024, (int)(CBS * 1024));
  k_headwise4_bf<<<cgrid(CBS * 1024), TB, 0, stream>>>(xc_buf, m_wq, qb, (int)(CBS * 1024));
  k_headwise4_bf<<<cgrid(CBS * 1024), TB, 0, stream>>>(xc_buf, m_wk, kb, (int)(CBS * 1024));
  k_headwise4_bf<<<cgrid(CBS * 1024), TB, 0, stream>>>(xm_buf, m_wv, vb, (int)(CBS * 1024));
  k_gates<<<dim3(M), 128, 0, stream>>>(qb, kb, vb, m_wig, m_big, m_wfg, m_bfg, ig_buf, fg_buf);
  k_mlstm_scan<<<dim3(1), 128, 0, stream>>>(ig_buf, fg_buf, lfc_buf, mrw_buf);
  k_mlstm_attn<<<dim3(CS / 16, CB * CNH_M), 32, 0, stream>>>(qb, kb, vb, ig_buf, lfc_buf, mrw_buf, att);
  k_mh_gate<<<dim3(M * 4), 256, 0, stream>>>(att, xc_buf, z_buf, m_outnorm, m_skip, gbf);
  gemm(gbf, 1024, wdown_bf, 1024, nullptr, h_buf, 512, h_buf, 512, M, 512, 1024);

  // ---------------- sLSTM block ----------------
  k_layernorm<<<dim3(M), 256, 0, stream>>>(h_buf, s_ln, ln_buf);
  conv_bf(ln_buf, lnbf, CBS * 512);
  k_conv_silu<<<cgrid(CBS * 512), TB, 0, stream>>>(ln_buf, s_conv_w, s_conv_b, xc2, 512, (int)(CBS * 512));
  conv_bf(xc2, xc2bf, CBS * 512);
  for (int h = 0; h < CNH_S; ++h) {
    for (int g = 0; g < 4; ++g) {
      const __bf16* Ain = (g < 2) ? xc2bf : lnbf;             // i,f from conv branch; z,o raw
      gemm(Ain + h * 128, 512, sw_bf[g] + (size_t)h * 128 * 128, 128,
           nullptr, nullptr, 0, wx_buf + (size_t)h * 512 + g * 128, 2048, M, 128, 128);
    }
  }
  k_slstm_scan<<<dim3(CB * CNH_S), 128, 0, stream>>>(wx_buf, s_R, s_bv, y_buf);
  k_slstm_norm_res<<<dim3(M * 4), 128, 0, stream>>>(y_buf, s_gn, h_buf);

  // ---------------- gated FFN ----------------
  k_layernorm<<<dim3(M), 256, 0, stream>>>(h_buf, s_ln2, ln_buf);
  conv_bf(ln_buf, lnbf, CBS * 512);
  gemm(lnbf, 512, fup_bf, 512, nullptr, nullptr, 0, ffup, 1408, M, 1408, 512);
  k_ffn_act<<<cgrid(CBS * CFF), TB, 0, stream>>>(ffup, ffbf, (int)(CBS * CFF));
  gemm(ffbf, CFF, fdn_bf, CFF, nullptr, h_buf, 512, h_buf, 512, M, 512, CFF);

  // ---------------- head ----------------
  k_layernorm<<<dim3(M), 256, 0, stream>>>(h_buf, post_ln, ln_buf);
  k_final<<<dim3(CB), 256, 0, stream>>>(ln_buf, w_fc, b_fc, out);
}
